// ResBlockNN_18425409700534
// MI455X (gfx1250) — matrix operand order repaired and checked by reading, not compiled, not run on hardware
//
#include <hip/hip_runtime.h>

// ---------------------------------------------------------------------------
// ResBlockNN on MI455X (gfx1250): bf16 WMMA gather-conv pipeline with
// TDM (tensor_load_to_lds) gather-mode DMA for the indexed gathers.
//   B=32, Cin=32, Cout=64, Nin=32768, Nout=8192, K=9, KP=4
// ---------------------------------------------------------------------------

typedef __attribute__((ext_vector_type(16))) __bf16 v16bf;
typedef __attribute__((ext_vector_type(8)))  float  v8f;
typedef __attribute__((ext_vector_type(4)))  unsigned int v4u;
typedef __attribute__((ext_vector_type(8)))  int    v8i;
typedef __attribute__((ext_vector_type(4)))  int    v4i;

union Frag { v16bf v; uint4 q[2]; };

__device__ __forceinline__ unsigned short f2bf(float f) {
  unsigned int u = __float_as_uint(f);
  u += 0x7FFFu + ((u >> 16) & 1u);   // round-to-nearest-even
  return (unsigned short)(u >> 16);
}
__device__ __forceinline__ float bf2f(unsigned short b) {
  return __uint_as_float((unsigned int)b << 16);
}

#define NB    32
#define CIN   32
#define COUT  64
#define NIN   32768
#define NOUT  8192
#define KW    9
#define KP    4
#define KG1   288   // CIN*KW
#define KG2   576   // COUT*KW
#define NPT   4     // output points per workgroup in the conv kernels
#define CNT_INV (1.0f / (32.0f * 8192.0f))

#if defined(__has_builtin)
#if __has_builtin(__builtin_amdgcn_tensor_load_to_lds) && \
    __has_builtin(__builtin_amdgcn_s_wait_tensorcnt)
#define HAVE_TDM 1
#endif
#endif

#ifdef HAVE_TDM
// ---------------------------------------------------------------------------
// TDM gather-mode DMA: gather NR indexed rows (each `dim0` bf16 elements)
// of a 2D tensor (nrowsTensor x dim0, bf16) into LDS at ldsAddr.
// Descriptor per CDNA5 ISA D# layout (group0/1 + 16-bit indices in groups
// 2/3). This toolchain's builtin takes 6 args: groups 0..3, an extra 8-dword
// group (unused for 2D gather -> zero), and cache policy.
// Must be executed by a single wave; completion via TENSORcnt.
// ---------------------------------------------------------------------------
template <int NR>
__device__ __forceinline__ void tdm_gather_rows(
    const unsigned short* base, unsigned ldsAddr, unsigned dim0,
    unsigned nrowsTensor, const int* __restrict__ rows) {
  unsigned long long ga = (unsigned long long)(uintptr_t)base;
  v4u g0;
  g0.x = 0x80000001u;                       // count=1, gather_mode=1, 16b idx
  g0.y = ldsAddr;                           // lds_addr
  g0.z = (unsigned)ga;                      // global_addr[31:0]
  g0.w = ((unsigned)(ga >> 32) & 0x1FFFFFFu) | (2u << 30);  // [56:32]|type=2
  v8i g1;
  g1[0] = 0x00010000;                       // data_size=1 (2 bytes)
  g1[1] = (int)((dim0 & 0xFFFFu) << 16);    // tensor_dim0[15:0] at bits 63:48
  g1[2] = (int)(((nrowsTensor & 0xFFFFu) << 16) | (dim0 >> 16));
  g1[3] = (int)(((dim0 & 0xFFFFu) << 16) | (nrowsTensor >> 16)); // tile_dim0
  g1[4] = NR;                               // tile_dim1 = #valid indices
  g1[5] = (int)dim0;                        // tensor_dim0_stride[31:0]
  g1[6] = 0;
  g1[7] = 0;
  unsigned w[5] = {0u, 0u, 0u, 0u, 0u};
#pragma unroll
  for (int j = 0; j < NR; ++j) {
    unsigned v = (unsigned)__builtin_amdgcn_readfirstlane(rows[j]) & 0xFFFFu;
    w[j >> 1] |= v << ((j & 1) * 16);
  }
  v4i g2 = {(int)w[0], (int)w[1], (int)w[2], (int)w[3]};
  v4i g3 = {(int)w[4], 0, 0, 0};
  v8i gx = {0, 0, 0, 0, 0, 0, 0, 0};        // extra group: unused (2D gather)
  __builtin_amdgcn_tensor_load_to_lds(g0, g1, g2, g3, gx, 0);
}
#else
// Fallback: vectorized gather copy (bf16 rows are 16B-aligned).
template <int NR>
__device__ __forceinline__ void copy_gather_rows(
    const unsigned short* base, unsigned short* dst, unsigned dim0,
    const int* __restrict__ rows, int tid) {
  const int chunks = (int)dim0 >> 3;          // uint4 chunks per row
  for (int c = tid; c < NR * chunks; c += 256) {
    int r = c / chunks, off = c % chunks;
    ((uint4*)dst)[c] =
        ((const uint4*)(base + (size_t)rows[r] * dim0))[off];
  }
}
#endif

// ---------------------------------------------------------------------------
// 1) Transpose x (B,Cin,Nin) fp32 -> xTbf[n][b][i] bf16 (contiguous 2KB/point)
// ---------------------------------------------------------------------------
__global__ __launch_bounds__(256) void transpose_kernel(
    const float* __restrict__ x, unsigned short* __restrict__ xTbf) {
  __shared__ float tile[32][33];
  const int pT = blockIdx.x * 32;
  const int cT = blockIdx.y * 32;
  const int tid = threadIdx.x;
  for (int l = tid; l < 1024; l += 256) {
    int cc = l >> 5, pp = l & 31;
    tile[cc][pp] = x[(size_t)(cT + cc) * NIN + pT + pp];       // coalesced p
  }
  __syncthreads();
  for (int l = tid; l < 1024; l += 256) {
    int pp = l >> 5, cc = l & 31;
    xTbf[(size_t)(pT + pp) * 1024 + cT + cc] = f2bf(tile[cc][pp]);
  }
}

// ---------------------------------------------------------------------------
// 2) Convert weights to bf16 with WMMA-friendly K ordering.
// ---------------------------------------------------------------------------
__global__ __launch_bounds__(256) void wconv_kernel(
    const float* __restrict__ W1, const float* __restrict__ W2,
    const float* __restrict__ We,
    unsigned short* __restrict__ W1bf, unsigned short* __restrict__ W2bf,
    unsigned short* __restrict__ Webf) {
  int t = blockIdx.x * 256 + threadIdx.x;
  if (t < COUT * KG1) {
    int o = t / KG1, r = t % KG1, k = r >> 5, i = r & 31;
    W1bf[t] = f2bf(W1[(o * CIN + i) * KW + k]);
  }
  if (t < COUT * KG2) {
    int o = t / KG2, r = t % KG2, k = r >> 6, oo = r & 63;
    W2bf[t] = f2bf(W2[(o * COUT + oo) * KW + k]);
  }
  if (t < COUT * CIN) Webf[t] = f2bf(We[t]);
}

// ---------------------------------------------------------------------------
// 3) conv1: TDM-gather 9 rows -> GTr[k][b][i]; 9 bf16 WMMA K-steps.
//    Writes z1T[n][b][o] (pre-BN fp32) + per-channel sum/sumsq.
// ---------------------------------------------------------------------------
__global__ __launch_bounds__(256) void conv1_kernel(
    const unsigned short* __restrict__ xTbf,
    const unsigned short* __restrict__ W1bf, const float* __restrict__ b1,
    const int* __restrict__ idx1, float* __restrict__ z1T,
    float* __restrict__ stat_s, float* __restrict__ stat_q) {
  __shared__ unsigned short GTr[KW][NB][CIN];      // [9][32][32] = 18KB
  __shared__ float sS[COUT], sQ[COUT], sB[COUT];
  const int tid  = threadIdx.x;
  const int lane = tid & 31, wave = tid >> 5;
  const int mtile = wave & 3, ntile = wave >> 2;   // 4 x 2 C tiles
  const int m = lane & 15, half = lane >> 4;
  if (tid < COUT) { sS[tid] = 0.f; sQ[tid] = 0.f; sB[tid] = b1[tid]; }
  __syncthreads();

  const int n0 = blockIdx.x * NPT;
  for (int np = 0; np < NPT; ++np) {
    const int n = n0 + np;
#ifdef HAVE_TDM
    if (wave == 0) {
      tdm_gather_rows<KW>(xTbf, (unsigned)(uintptr_t)(void*)&GTr[0][0][0],
                          1024u, (unsigned)NIN, idx1 + n * KW);
      __builtin_amdgcn_s_wait_tensorcnt(0);
    }
#else
    copy_gather_rows<KW>(xTbf, &GTr[0][0][0], 1024u, idx1 + n * KW, tid);
#endif
    __syncthreads();

    v8f c = {};
    const unsigned short* wrow = W1bf + (size_t)(mtile * 16 + m) * KG1;
#pragma unroll
    for (int kk = 0; kk < KW; ++kk) {
      Frag a, b;
      const uint4* wp = (const uint4*)(wrow + kk * 32 + half * 8);
      a.q[0] = wp[0];                       // K = kb+k0 .. +7
      a.q[1] = wp[2];                       // K = kb+k0+16 .. +23
      const uint4* gp = (const uint4*)(&GTr[kk][ntile * 16 + m][half * 8]);
      b.q[0] = gp[0]; b.q[1] = gp[2];
      c = __builtin_amdgcn_wmma_f32_16x16x32_bf16(false, a.v, false, b.v,
                                                  (short)0, c, false, false);
    }
    // C layout: VGPR r -> (o = mtile*16 + 8*half + r, b = ntile*16 + m)
    const int obase = mtile * 16 + 8 * half;
    v8f vout;
#pragma unroll
    for (int r = 0; r < 8; ++r) {
      float v = c[r] + sB[obase + r];
      vout[r] = v;
      atomicAdd(&sS[obase + r], v);
      atomicAdd(&sQ[obase + r], v * v);
    }
    *(v8f*)(z1T + ((size_t)n * NB + ntile * 16 + m) * COUT + obase) = vout;
    __syncthreads();
  }
  if (tid < COUT) {
    atomicAdd(&stat_s[tid], sS[tid]);
    atomicAdd(&stat_q[tid], sQ[tid]);
  }
}

// ---------------------------------------------------------------------------
// 4) Finalize BN stats -> fused scale/shift
// ---------------------------------------------------------------------------
__global__ void finalize_kernel(const float* __restrict__ s,
                                const float* __restrict__ q,
                                const float* __restrict__ g,
                                const float* __restrict__ beta,
                                float* __restrict__ scale,
                                float* __restrict__ shift) {
  int o = threadIdx.x;
  if (o < COUT) {
    float mean = s[o] * CNT_INV;
    float var  = q[o] * CNT_INV - mean * mean;
    float sc   = g[o] * rsqrtf(var + 1e-5f);
    scale[o] = sc;
    shift[o] = beta[o] - mean * sc;
  }
}

// ---------------------------------------------------------------------------
// 5) BN1 + ReLU elementwise: z1T fp32 -> act1 bf16 (TDM-copyable rows)
// ---------------------------------------------------------------------------
__global__ __launch_bounds__(256) void bnrelu_kernel(
    const float* __restrict__ z1T, const float* __restrict__ sc1,
    const float* __restrict__ sh1, unsigned short* __restrict__ act1) {
  size_t j0 = ((size_t)blockIdx.x * 256 + threadIdx.x) * 8;
  v8f v = *(const v8f*)(z1T + j0);
  int ob = (int)(j0 & 63);
  union { unsigned short u[8]; uint4 q; } r;
#pragma unroll
  for (int t = 0; t < 8; ++t)
    r.u[t] = f2bf(fmaxf(v[t] * sc1[ob + t] + sh1[ob + t], 0.0f));
  *(uint4*)(act1 + j0) = r.q;
}

// ---------------------------------------------------------------------------
// 6) conv2 (gathers act1 bf16 via TDM) + identity branch (TDM pool gather).
//    Writes z2T[n][b][o], zidT[n][b][o]; accumulates stats2 & stats3.
// ---------------------------------------------------------------------------
__global__ __launch_bounds__(256) void conv2id_kernel(
    const unsigned short* __restrict__ xTbf,
    const unsigned short* __restrict__ act1,
    const unsigned short* __restrict__ W2bf,
    const unsigned short* __restrict__ Webf,
    const float* __restrict__ b2, const float* __restrict__ be,
    const int* __restrict__ idx2, const int* __restrict__ pool_idx,
    float* __restrict__ z2T, float* __restrict__ zidT,
    float* __restrict__ s2, float* __restrict__ q2,
    float* __restrict__ s3, float* __restrict__ q3) {
  __shared__ unsigned short GTr[KW][NB][COUT];     // [9][32][64] = 36KB
  __shared__ unsigned short PPr[KP][NB][CIN];      // [4][32][32] = 8KB
  __shared__ unsigned short PP[NB][CIN];           // pooled bf16, 2KB
  __shared__ float sS2[COUT], sQ2[COUT], sS3[COUT], sQ3[COUT];
  __shared__ float sB2[COUT], sBe[COUT];
  const int tid  = threadIdx.x;
  const int lane = tid & 31, wave = tid >> 5;
  const int mtile = wave & 3, ntile = wave >> 2;
  const int m = lane & 15, half = lane >> 4;
  if (tid < COUT) {
    sS2[tid] = 0.f; sQ2[tid] = 0.f; sS3[tid] = 0.f; sQ3[tid] = 0.f;
    sB2[tid] = b2[tid]; sBe[tid] = be[tid];
  }
  __syncthreads();

  const int n0 = blockIdx.x * NPT;
  for (int np = 0; np < NPT; ++np) {
    const int n = n0 + np;
#ifdef HAVE_TDM
    if (wave == 0) {
      tdm_gather_rows<KW>(act1, (unsigned)(uintptr_t)(void*)&GTr[0][0][0],
                          2048u, (unsigned)NOUT, idx2 + n * KW);
      tdm_gather_rows<KP>(xTbf, (unsigned)(uintptr_t)(void*)&PPr[0][0][0],
                          1024u, (unsigned)NIN, pool_idx + n * KP);
      __builtin_amdgcn_s_wait_tensorcnt(0);
    }
#else
    copy_gather_rows<KW>(act1, &GTr[0][0][0], 2048u, idx2 + n * KW, tid);
    copy_gather_rows<KP>(xTbf, &PPr[0][0][0], 1024u, pool_idx + n * KP, tid);
#endif
    __syncthreads();

    // pooled = mean of the 4 gathered rows (bf16 in, bf16 out)
    for (int e = tid; e < 1024; e += 256) {
      int b = e >> 5, i = e & 31;
      float s = bf2f(PPr[0][b][i]) + bf2f(PPr[1][b][i]) +
                bf2f(PPr[2][b][i]) + bf2f(PPr[3][b][i]);
      PP[b][i] = f2bf(s * 0.25f);
    }
    __syncthreads();

    // conv2: 18 bf16 WMMA K-steps over GTr[k][b][c] (K = k*64 + c)
    v8f c = {};
    const unsigned short* wrow = W2bf + (size_t)(mtile * 16 + m) * KG2;
#pragma unroll
    for (int kk = 0; kk < 2 * KW; ++kk) {
      Frag a, b;
      const uint4* wp = (const uint4*)(wrow + kk * 32 + half * 8);
      a.q[0] = wp[0];
      a.q[1] = wp[2];
      const uint4* gp = (const uint4*)(
          &GTr[kk >> 1][ntile * 16 + m][(kk & 1) * 32 + half * 8]);
      b.q[0] = gp[0]; b.q[1] = gp[2];
      c = __builtin_amdgcn_wmma_f32_16x16x32_bf16(false, a.v, false, b.v,
                                                  (short)0, c, false, false);
    }
    // identity: single WMMA step (K = 32)
    v8f ci = {};
    {
      Frag a, b;
      const uint4* wp = (const uint4*)(Webf + (size_t)(mtile * 16 + m) * CIN +
                                       half * 8);
      a.q[0] = wp[0];
      a.q[1] = wp[2];
      const uint4* gp = (const uint4*)(&PP[ntile * 16 + m][half * 8]);
      b.q[0] = gp[0]; b.q[1] = gp[2];
      ci = __builtin_amdgcn_wmma_f32_16x16x32_bf16(false, a.v, false, b.v,
                                                   (short)0, ci, false, false);
    }

    const int obase = mtile * 16 + 8 * half;
    const size_t dst = ((size_t)n * NB + ntile * 16 + m) * COUT + obase;
    v8f v2, vi;
#pragma unroll
    for (int r = 0; r < 8; ++r) {
      float a2 = c[r]  + sB2[obase + r];
      float ai = ci[r] + sBe[obase + r];
      v2[r] = a2; vi[r] = ai;
      atomicAdd(&sS2[obase + r], a2);
      atomicAdd(&sQ2[obase + r], a2 * a2);
      atomicAdd(&sS3[obase + r], ai);
      atomicAdd(&sQ3[obase + r], ai * ai);
    }
    *(v8f*)(z2T + dst)  = v2;
    *(v8f*)(zidT + dst) = vi;
    __syncthreads();
  }
  if (tid < COUT) {
    atomicAdd(&s2[tid], sS2[tid]);
    atomicAdd(&q2[tid], sQ2[tid]);
    atomicAdd(&s3[tid], sS3[tid]);
    atomicAdd(&q3[tid], sQ3[tid]);
  }
}

// ---------------------------------------------------------------------------
// 7) out = relu(bn2(z2) + bn3(zid)), LDS transpose to (B,Cout,N) layout
// ---------------------------------------------------------------------------
__global__ __launch_bounds__(256) void final_kernel(
    const float* __restrict__ z2T, const float* __restrict__ zidT,
    const float* __restrict__ sc2, const float* __restrict__ sh2,
    const float* __restrict__ sc3, const float* __restrict__ sh3,
    float* __restrict__ out) {
  __shared__ float t[32][65];                 // [bo][n]
  const int nbase  = blockIdx.x * 64;
  const int bobase = blockIdx.y * 32;
  const int tid = threadIdx.x;
  for (int l = tid; l < 2048; l += 256) {
    int nn = l >> 5, bb = l & 31;
    size_t src = (size_t)(nbase + nn) * 2048 + bobase + bb;  // coalesced bb
    int o = (bobase + bb) & 63;
    float v = z2T[src] * sc2[o] + sh2[o] + zidT[src] * sc3[o] + sh3[o];
    t[bb][nn] = fmaxf(v, 0.0f);
  }
  __syncthreads();
  for (int l = tid; l < 2048; l += 256) {
    int bb = l >> 6, nn = l & 63;
    out[(size_t)(bobase + bb) * NOUT + nbase + nn] = t[bb][nn];  // coalesced
  }
}

// ---------------------------------------------------------------------------
// Workspace layout (requires ws_size >= ~288 MB)
// ---------------------------------------------------------------------------
static constexpr size_t OFF_XT   = 0;                                // 64 MB
static constexpr size_t OFF_Z1   = OFF_XT  + (size_t)NIN * 1024 * 2;  // 64 MB
static constexpr size_t OFF_ACT1 = OFF_Z1  + (size_t)NOUT * 2048 * 4; // 32 MB
static constexpr size_t OFF_Z2   = OFF_ACT1 + (size_t)NOUT * 2048 * 2;
static constexpr size_t OFF_ZID  = OFF_Z2  + (size_t)NOUT * 2048 * 4;
static constexpr size_t OFF_W1B  = OFF_ZID + (size_t)NOUT * 2048 * 4;
static constexpr size_t OFF_W2B  = OFF_W1B + (size_t)COUT * KG1 * 2;
static constexpr size_t OFF_WEB  = OFF_W2B + (size_t)COUT * KG2 * 2;
static constexpr size_t OFF_STAT = OFF_WEB + (size_t)COUT * CIN * 2;
static constexpr size_t OFF_SC   = OFF_STAT + 6 * COUT * 4;

extern "C" void kernel_launch(void* const* d_in, const int* in_sizes, int n_in,
                              void* d_out, int out_size, void* d_ws,
                              size_t ws_size, hipStream_t stream) {
  const float* x     = (const float*)d_in[0];
  const float* W1    = (const float*)d_in[1];
  const float* b1    = (const float*)d_in[2];
  const float* W2    = (const float*)d_in[3];
  const float* b2    = (const float*)d_in[4];
  const float* We    = (const float*)d_in[5];
  const float* be    = (const float*)d_in[6];
  const float* g1    = (const float*)d_in[7];
  const float* beta1 = (const float*)d_in[8];
  const float* g2    = (const float*)d_in[9];
  const float* beta2 = (const float*)d_in[10];
  const float* g3    = (const float*)d_in[11];
  const float* beta3 = (const float*)d_in[12];
  const int* idx1    = (const int*)d_in[13];
  const int* idx2    = (const int*)d_in[14];
  const int* pidx    = (const int*)d_in[15];
  float* out = (float*)d_out;

  char* ws = (char*)d_ws;
  unsigned short* xTbf = (unsigned short*)(ws + OFF_XT);
  float*          z1T  = (float*)(ws + OFF_Z1);
  unsigned short* act1 = (unsigned short*)(ws + OFF_ACT1);
  float*          z2T  = (float*)(ws + OFF_Z2);
  float*          zidT = (float*)(ws + OFF_ZID);
  unsigned short* W1bf = (unsigned short*)(ws + OFF_W1B);
  unsigned short* W2bf = (unsigned short*)(ws + OFF_W2B);
  unsigned short* Webf = (unsigned short*)(ws + OFF_WEB);
  float* st = (float*)(ws + OFF_STAT);
  float* s1 = st + 0 * COUT, *q1 = st + 1 * COUT;
  float* s2 = st + 2 * COUT, *q2 = st + 3 * COUT;
  float* s3 = st + 4 * COUT, *q3 = st + 5 * COUT;
  float* sc = (float*)(ws + OFF_SC);
  float* sc1 = sc + 0 * COUT, *sh1 = sc + 1 * COUT;
  float* sc2 = sc + 2 * COUT, *sh2 = sc + 3 * COUT;
  float* sc3 = sc + 4 * COUT, *sh3 = sc + 5 * COUT;

  (void)hipMemsetAsync(st, 0, 6 * COUT * sizeof(float), stream);

  wconv_kernel<<<(COUT * KG2 + 255) / 256, 256, 0, stream>>>(W1, W2, We, W1bf,
                                                             W2bf, Webf);
  transpose_kernel<<<dim3(NIN / 32, 1024 / 32), 256, 0, stream>>>(x, xTbf);

  conv1_kernel<<<NOUT / NPT, 256, 0, stream>>>(xTbf, W1bf, b1, idx1, z1T, s1,
                                               q1);
  finalize_kernel<<<1, 64, 0, stream>>>(s1, q1, g1, beta1, sc1, sh1);
  bnrelu_kernel<<<(NOUT * 2048) / (256 * 8), 256, 0, stream>>>(z1T, sc1, sh1,
                                                               act1);

  conv2id_kernel<<<NOUT / NPT, 256, 0, stream>>>(xTbf, act1, W2bf, Webf, b2,
                                                 be, idx2, pidx, z2T, zidT,
                                                 s2, q2, s3, q3);
  finalize_kernel<<<1, 64, 0, stream>>>(s2, q2, g2, beta2, sc2, sh2);
  finalize_kernel<<<1, 64, 0, stream>>>(s3, q3, g3, beta3, sc3, sh3);

  final_kernel<<<dim3(NOUT / 64, 2048 / 32), 256, 0, stream>>>(
      z2T, zidT, sc2, sh2, sc3, sh3, out);
}
